// MIMOJacobiSSM_ND_77043123355943
// MI455X (gfx1250) — matrix operand order repaired and checked by reading, not compiled, not run on hardware
//
#include <hip/hip_runtime.h>
#include <hip/hip_bf16.h>

// ---------------------------------------------------------------------------
// MIMO Jacobi SSM, CDNA5 (gfx1250) implementation.
//  - Projection + output GEMMs via v_wmma_f32_16x16x32_f16 (f32 accumulate)
//  - Weight tiles staged to LDS via async global->LDS (ASYNCcnt) and read
//    through ds_load; A fragments reused across multiple N tiles per wave.
//  - 12 sequential stencil/update steps (memory bound, L2 resident)
// ---------------------------------------------------------------------------

typedef __attribute__((ext_vector_type(16))) _Float16 v16h;
typedef __attribute__((ext_vector_type(8)))  _Float16 v8h;
typedef __attribute__((ext_vector_type(8)))  float    v8f;

#define BSZ    16
#define HS     48
#define WSZ    48
#define DDIM   768
#define NCH    64
#define RR     4
#define NITER  12
#define NRTOT  256
#define PIX    (BSZ*HS*WSZ)            // 36864
#define TOT    (PIX*RR*NCH)            // 9437184 elements of state
#define NTPROJ 39                      // valid column tiles: 16 B, 16 C, 4 dec, 2 th, 1 {X,lam}
#define NTPAD  40                      // padded to even count for tile-pairing
#define TGP    2                       // proj: weight tiles per block
#define TGO    4                       // out:  weight tiles per block

__device__ __forceinline__ float siluf(float v) { return v / (1.f + __expf(-v)); }
__device__ __forceinline__ float sigmf(float v) { return 1.f / (1.f + __expf(-v)); }

// Async global->LDS copy of one 16-byte chunk (per-lane), gfx1250 ASYNC path.
__device__ __forceinline__ void async_copy_b128(unsigned lds_byte_addr, const void* gaddr)
{
    asm volatile("global_load_async_to_lds_b128 %0, %1, off"
                 :: "v"(lds_byte_addr), "v"(gaddr) : "memory");
}
__device__ __forceinline__ void wait_async0()
{
    asm volatile("s_wait_asynccnt 0" ::: "memory");
}

// A-fragment load (row-major [16,K] f16, lda elems/row) in contiguous k order (same order as the B frag):
//   lane L: M = L&15; elems e: K = e + 16*(L>>4)
__device__ __forceinline__ v16h load_a_frag(const _Float16* __restrict__ arow, int k0, int half)
{
    v8h alo = *(const v8h*)(arow + k0 + 16 * half);
    v8h ahi = *(const v8h*)(arow + k0 + 8 + 16 * half);
    return __builtin_shufflevector(alo, ahi, 0,1,2,3,4,5,6,7,8,9,10,11,12,13,14,15);
}

// ------------------------------ prep kernels -------------------------------

__global__ __launch_bounds__(256) void cvt_x_kernel(const float* __restrict__ x,
                                                    _Float16* __restrict__ x16)
{
    size_t i = (size_t)blockIdx.x * 256 + threadIdx.x;   // PIX*DDIM total
    x16[i] = (_Float16)x[i];
}

// Pack all projection weights transposed into Wt[640][768] (row = output col),
// rows 624..639 zero (padding tile 39).
__global__ __launch_bounds__(256) void pack_wcat_kernel(
    const float* __restrict__ W_B, const float* __restrict__ W_C,
    const float* __restrict__ W_dec, const float* __restrict__ W_th,
    const float* __restrict__ W_X, const float* __restrict__ W_lam,
    _Float16* __restrict__ Wt)
{
    size_t i = (size_t)blockIdx.x * 256 + threadIdx.x;   // 640*768 total
    int rr = (int)(i / DDIM);      // packed output column 0..639
    int k  = (int)(i % DDIM);
    float v = 0.f;
    if (rr < 256)       v = W_B[(size_t)k * 256 + rr];
    else if (rr < 512)  v = W_C[(size_t)k * 256 + (rr - 256)];
    else if (rr < 576)  v = W_dec[(size_t)k * 64 + (rr - 512)];
    else if (rr < 608)  v = W_th[(size_t)k * 32 + (rr - 576)];
    else if (rr < 624) {
        int j = rr - 608;          // tile 38: cols 0-3 = W_X, col 4 = W_lam
        v = (j < 4) ? W_X[(size_t)k * 4 + j] : (j == 4 ? W_lam[k] : 0.f);
    }
    Wt[i] = (_Float16)v;
}

__global__ __launch_bounds__(256) void pack_bias_kernel(
    const float* __restrict__ b_B, const float* __restrict__ B_bias,
    const float* __restrict__ b_C, const float* __restrict__ C_bias,
    const float* __restrict__ b_dec, const float* __restrict__ b_th,
    const float* __restrict__ b_X, const float* __restrict__ b_lam,
    float* __restrict__ bias_all)
{
    int i = blockIdx.x * 256 + threadIdx.x;
    if (i >= NTPAD * 16) return;
    float v = 0.f;
    if (i < 256)       v = b_B[i] + B_bias[i];
    else if (i < 512)  v = b_C[i - 256] + C_bias[i - 256];
    else if (i < 576)  v = b_dec[i - 512];
    else if (i < 608)  v = b_th[i - 576];
    else if (i < 624) {
        int j = i - 608;
        v = (j < 4) ? b_X[j] : (j == 4 ? b_lam[0] : 0.f);
    }
    bias_all[i] = v;
}

__global__ __launch_bounds__(256) void pack_wout_kernel(const float* __restrict__ W_out,
                                                        _Float16* __restrict__ Wt_out)
{
    size_t i = (size_t)blockIdx.x * 256 + threadIdx.x;   // 768*256 total
    int n = (int)(i / NRTOT);
    int k = (int)(i % NRTOT);
    Wt_out[i] = (_Float16)W_out[(size_t)k * DDIM + n];   // W_out[256,768] -> Wt[768,256]
}

// ------------------------- fused projection GEMM ---------------------------
// grid = (PIX/128, NTPAD/TGP), 256 threads = 8 waves; wave w: rows M0=bx*128+16w.
// Each block stages TGP weight tiles (TGP*16 rows x 768) in LDS via async copy;
// each wave keeps TGP accumulators and reuses its A fragment TGP times.
__global__ __launch_bounds__(256) void proj_gemm_kernel(
    const _Float16* __restrict__ x16, const _Float16* __restrict__ Wt,
    const float* __restrict__ bias_all,
    float* __restrict__ Bp, float* __restrict__ Cp,
    float* __restrict__ alphaB, float* __restrict__ thetaB,
    float* __restrict__ XLB)
{
    __shared__ __attribute__((aligned(64))) _Float16 smemB[TGP * 16 * DDIM]; // 48 KB

    const int wave = threadIdx.x >> 5;
    const int lane = threadIdx.x & 31;
    const int M0   = blockIdx.x * 128 + wave * 16;
    const int t0   = blockIdx.y * TGP;

    // cooperative async load of TGP weight tiles into LDS (16B chunks)
    {
        const char* gB = (const char*)(Wt + (size_t)t0 * 16 * DDIM);
        unsigned base  = (unsigned)(size_t)&smemB[0];
        const int nchunks = TGP * 16 * DDIM * 2 / 16;   // 3072
        for (int c = threadIdx.x; c < nchunks; c += 256)
            async_copy_b128(base + (unsigned)c * 16u, gB + (size_t)c * 16);
        wait_async0();
        __syncthreads();
    }

    const int row  = lane & 15;
    const int half = lane >> 4;
    const _Float16* arow = x16 + (size_t)(M0 + row) * DDIM;

    v8f acc[TGP];
#pragma unroll
    for (int j = 0; j < TGP; ++j) acc[j] = (v8f){};

    for (int k0 = 0; k0 < DDIM; k0 += 32) {
        v16h a = load_a_frag(arow, k0, half);
#pragma unroll
        for (int j = 0; j < TGP; ++j) {
            // B frag: lane L -> N = L&15, elems = 16 contiguous K at k0+16*half
            v16h b = *(const v16h*)(&smemB[(size_t)(j * 16 + row) * DDIM + k0 + 16 * half]);
            acc[j] = __builtin_amdgcn_wmma_f32_16x16x32_f16(
                         false, a, false, b, (short)0, acc[j], false, false);
        }
    }

#pragma unroll
    for (int j = 0; j < TGP; ++j) {
        int t = t0 + j;
        if (t >= NTPROJ) continue;                      // padding tile
        const float bias = bias_all[t * 16 + row];
#pragma unroll
        for (int v = 0; v < 8; ++v) {
            int m = M0 + v + 8 * half;                  // D: VGPR v -> M = v + 8*half
            float val = acc[j][v] + bias;
            if (t < 16) {
                Bp[(size_t)m * NRTOT + t * 16 + row] = siluf(val);
            } else if (t < 32) {
                Cp[(size_t)m * NRTOT + (t - 16) * 16 + row] = siluf(val);
            } else if (t < 36) {
                alphaB[(size_t)m * NCH + (t - 32) * 16 + row] = sigmf(val);
            } else if (t < 38) {
                thetaB[(size_t)m * 32 + (t - 36) * 16 + row] = val;
            } else {
                float o = (row < 4) ? siluf(val) : (row == 4 ? sigmf(val) : 0.f);
                XLB[(size_t)m * 16 + row] = o;  // cols 0-3: Xb(silu), col 4: gamma(sigmoid)
            }
        }
    }
}

// ------------------------------ Jacobi step --------------------------------
// Hst_new = alpha*depthwise3x3(Hst) + [i==0 ? inj_i : (1-g)*alpha*inj_{i-1} + g*inj_i]
// inj_i = rope(B, theta*(i+1)) * Xb   (recomputed; pure function of i)
__global__ __launch_bounds__(256) void step_kernel(
    const float* __restrict__ src, float* __restrict__ dst,
    const float* __restrict__ Bp, const float* __restrict__ thetaB,
    const float* __restrict__ alphaB, const float* __restrict__ XLB,
    const float* __restrict__ conv_w, const float* __restrict__ conv_b,
    int iter)
{
    int e = blockIdx.x * 256 + threadIdx.x;          // TOT total
    int n = e & (NCH - 1);
    int w = (e / NCH) % WSZ;
    int h = (e / (NCH * WSZ)) % HS;
    int r = (e / (NCH * WSZ * HS)) % RR;
    int b = e / (NCH * WSZ * HS * RR);
    int p = (b * HS + h) * WSZ + w;

    // depthwise 3x3 conv, SAME zero padding (input of iter 0 is all-zero state)
    float convAcc = conv_b[n];
    if (iter != 0) {
        int base = ((b * RR + r) * HS) * WSZ;        // (b,r) plane
#pragma unroll
        for (int dy = 0; dy < 3; ++dy) {
            int hy = h + dy - 1;
            if (hy < 0 || hy >= HS) continue;
#pragma unroll
            for (int dx = 0; dx < 3; ++dx) {
                int wx = w + dx - 1;
                if (wx < 0 || wx >= WSZ) continue;
                convAcc += src[(size_t)(base + hy * WSZ + wx) * NCH + n]
                           * conv_w[(dy * 3 + dx) * NCH + n];
            }
        }
    }

    // rope-modulated injection (pairs share theta[k])
    int   k   = n >> 1;
    float th  = thetaB[(size_t)p * 32 + k];
    float v1  = Bp[(size_t)p * NRTOT + (2 * k) * RR + r];
    float v2  = Bp[(size_t)p * NRTOT + (2 * k + 1) * RR + r];
    float xb  = XLB[(size_t)p * 16 + r];
    float s1, c1;
    __sincosf(th * (float)(iter + 1), &s1, &c1);
    float injCur = ((n & 1) ? (v1 * s1 + v2 * c1) : (v1 * c1 - v2 * s1)) * xb;

    float a = alphaB[(size_t)p * NCH + n];
    float out;
    if (iter == 0) {
        out = a * convAcc + injCur;
    } else {
        float s0, c0;
        __sincosf(th * (float)iter, &s0, &c0);
        float injPrev = ((n & 1) ? (v1 * s0 + v2 * c0) : (v1 * c0 - v2 * s0)) * xb;
        float g = XLB[(size_t)p * 16 + 4];
        out = a * convAcc + (1.f - g) * a * injPrev + g * injCur;
    }
    dst[e] = out;
}

// -------------------- gate by rope(C, 12*theta) -> f16 Ho ------------------
__global__ __launch_bounds__(256) void hg_kernel(
    const float* __restrict__ Hst, const float* __restrict__ Cp,
    const float* __restrict__ thetaB, _Float16* __restrict__ Ho16)
{
    int e = blockIdx.x * 256 + threadIdx.x;
    int n = e & (NCH - 1);
    int w = (e / NCH) % WSZ;
    int h = (e / (NCH * WSZ)) % HS;
    int r = (e / (NCH * WSZ * HS)) % RR;
    int b = e / (NCH * WSZ * HS * RR);
    int p = (b * HS + h) * WSZ + w;

    int   k  = n >> 1;
    float th = thetaB[(size_t)p * 32 + k] * (float)NITER;
    float s, c;
    __sincosf(th, &s, &c);
    float v1 = Cp[(size_t)p * NRTOT + (2 * k) * RR + r];
    float v2 = Cp[(size_t)p * NRTOT + (2 * k + 1) * RR + r];
    float crot = (n & 1) ? (v1 * s + v2 * c) : (v1 * c - v2 * s);
    // Ho[p, r*64 + n] (transpose (b,r,h,w,n)->(b,h,w,r,n))
    Ho16[(size_t)p * NRTOT + r * NCH + n] = (_Float16)(crot * Hst[e]);
}

// ------------------------------ output GEMM --------------------------------
// out = silu(Ho[36864,256] @ W_out[256,768] + b_out); grid (PIX/128, 768/(16*TGO))
__global__ __launch_bounds__(256) void out_gemm_kernel(
    const _Float16* __restrict__ Ho16, const _Float16* __restrict__ Wt_out,
    const float* __restrict__ b_out, float* __restrict__ out)
{
    __shared__ __attribute__((aligned(64))) _Float16 smemB[TGO * 16 * NRTOT]; // 32 KB

    const int wave = threadIdx.x >> 5;
    const int lane = threadIdx.x & 31;
    const int M0   = blockIdx.x * 128 + wave * 16;
    const int N0   = blockIdx.y * 16 * TGO;

    {
        const char* gB = (const char*)(Wt_out + (size_t)N0 * NRTOT);
        unsigned base  = (unsigned)(size_t)&smemB[0];
        const int nchunks = TGO * 16 * NRTOT * 2 / 16;  // 2048
        for (int c = threadIdx.x; c < nchunks; c += 256)
            async_copy_b128(base + (unsigned)c * 16u, gB + (size_t)c * 16);
        wait_async0();
        __syncthreads();
    }

    const int row  = lane & 15;
    const int half = lane >> 4;
    const _Float16* arow = Ho16 + (size_t)(M0 + row) * NRTOT;

    v8f acc[TGO];
#pragma unroll
    for (int j = 0; j < TGO; ++j) acc[j] = (v8f){};

    for (int k0 = 0; k0 < NRTOT; k0 += 32) {
        v16h a = load_a_frag(arow, k0, half);
#pragma unroll
        for (int j = 0; j < TGO; ++j) {
            v16h b = *(const v16h*)(&smemB[(size_t)(j * 16 + row) * NRTOT + k0 + 16 * half]);
            acc[j] = __builtin_amdgcn_wmma_f32_16x16x32_f16(
                         false, a, false, b, (short)0, acc[j], false, false);
        }
    }

#pragma unroll
    for (int j = 0; j < TGO; ++j) {
        const int nc = N0 + j * 16 + row;
        const float bias = b_out[nc];
#pragma unroll
        for (int v = 0; v < 8; ++v) {
            int m = M0 + v + 8 * half;
            out[(size_t)m * DDIM + nc] = siluf(acc[j][v] + bias);
        }
    }
}

// ------------------------------- launcher ----------------------------------

extern "C" void kernel_launch(void* const* d_in, const int* in_sizes, int n_in,
                              void* d_out, int out_size, void* d_ws, size_t ws_size,
                              hipStream_t stream)
{
    (void)in_sizes; (void)n_in; (void)out_size; (void)ws_size;
    const float* x      = (const float*)d_in[0];
    const float* W_B    = (const float*)d_in[1];
    const float* b_B    = (const float*)d_in[2];
    const float* W_C    = (const float*)d_in[3];
    const float* b_C    = (const float*)d_in[4];
    const float* W_X    = (const float*)d_in[5];
    const float* b_X    = (const float*)d_in[6];
    const float* W_dec  = (const float*)d_in[7];
    const float* b_dec  = (const float*)d_in[8];
    const float* W_th   = (const float*)d_in[9];
    const float* b_th   = (const float*)d_in[10];
    const float* W_lam  = (const float*)d_in[11];
    const float* b_lam  = (const float*)d_in[12];
    const float* B_bias = (const float*)d_in[13];
    const float* C_bias = (const float*)d_in[14];
    const float* conv_w = (const float*)d_in[15];
    const float* conv_b = (const float*)d_in[16];
    const float* W_out  = (const float*)d_in[17];
    const float* b_out  = (const float*)d_in[18];
    float* out = (float*)d_out;

    char* ws = (char*)d_ws;
    size_t off = 0;
    auto alloc = [&](size_t bytes) -> char* {
        char* p = ws + off;
        off = (off + bytes + 255) & ~(size_t)255;
        return p;
    };
    _Float16* x16      = (_Float16*)alloc((size_t)PIX * DDIM * 2);
    _Float16* Wt_cat   = (_Float16*)alloc((size_t)NTPAD * 16 * DDIM * 2);
    float*    bias_all = (float*)   alloc((size_t)NTPAD * 16 * 4);
    _Float16* Wt_out   = (_Float16*)alloc((size_t)DDIM * NRTOT * 2);
    float*    Bp       = (float*)   alloc((size_t)PIX * NRTOT * 4);
    float*    Cp       = (float*)   alloc((size_t)PIX * NRTOT * 4);
    float*    alphaB   = (float*)   alloc((size_t)PIX * NCH * 4);
    float*    thetaB   = (float*)   alloc((size_t)PIX * 32 * 4);
    float*    XLB      = (float*)   alloc((size_t)PIX * 16 * 4);
    float*    HstA     = (float*)   alloc((size_t)TOT * 4);
    float*    HstB     = (float*)   alloc((size_t)TOT * 4);
    _Float16* Ho16     = (_Float16*)alloc((size_t)PIX * NRTOT * 2);

    // prep
    cvt_x_kernel<<<(PIX * DDIM) / 256, 256, 0, stream>>>(x, x16);
    pack_wcat_kernel<<<(NTPAD * 16 * DDIM) / 256, 256, 0, stream>>>(
        W_B, W_C, W_dec, W_th, W_X, W_lam, Wt_cat);
    pack_bias_kernel<<<3, 256, 0, stream>>>(
        b_B, B_bias, b_C, C_bias, b_dec, b_th, b_X, b_lam, bias_all);
    pack_wout_kernel<<<(DDIM * NRTOT) / 256, 256, 0, stream>>>(W_out, Wt_out);

    // fused projections (WMMA, LDS-staged weights)
    dim3 pg(PIX / 128, NTPAD / TGP);
    proj_gemm_kernel<<<pg, 256, 0, stream>>>(x16, Wt_cat, bias_all,
                                             Bp, Cp, alphaB, thetaB, XLB);

    // 12 Jacobi steps, ping-pong buffers (dst = buf[i&1], src = buf[(i+1)&1])
    float* bufs[2] = { HstA, HstB };
    for (int i = 0; i < NITER; ++i) {
        step_kernel<<<TOT / 256, 256, 0, stream>>>(
            bufs[(i + 1) & 1], bufs[i & 1],
            Bp, thetaB, alphaB, XLB, conv_w, conv_b, i);
    }
    float* HstFinal = bufs[(NITER - 1) & 1];

    // gate + pack to f16
    hg_kernel<<<TOT / 256, 256, 0, stream>>>(HstFinal, Cp, thetaB, Ho16);

    // output GEMM (WMMA, LDS-staged weights) + silu
    dim3 og(PIX / 128, DDIM / (16 * TGO));
    out_gemm_kernel<<<og, 256, 0, stream>>>(Ho16, Wt_out, b_out, out);
}
